// ChildSumTreeLSTM_29154238005589
// MI455X (gfx1250) — matrix operand order repaired and checked by reading, not compiled, not run on hardware
//
#include <hip/hip_runtime.h>

typedef _Float16 v16h __attribute__((ext_vector_type(16)));
typedef float    v8f  __attribute__((ext_vector_type(8)));
typedef int      v4i  __attribute__((vector_size(16)));

#define T_DIM   1024
#define LEAVES  256
#define IN_DIM  300
#define MEM     150

// leaf GEMM: N = 450 -> 512 (32 tiles, 4/wave), K = 300 -> 320 (10 chunks of 32)
#define L_N      450
#define L_TILES  32
#define L_PITCH  512
#define L_KT     10
// level GEMM: N = 600 -> 640 (40 tiles, 5/wave), K = 150 -> 160 (5 chunks of 32)
#define V_N      600
#define V_TILES  40
#define V_PITCH  640
#define V_KT     5

#define WL_ELEMS (L_KT * L_TILES * 512)   // 163840 halves, frag-ready W_ioux^T
#define WV_ELEMS (V_KT * V_TILES * 512)   // 102400 halves, frag-ready [W_iouh^T|W_fh^T]

__device__ __forceinline__ float sigmoidf_(float x) {
    return 1.0f / (1.0f + __expf(-x));
}

// ---------------------------------------------------------------------
// Async global->LDS 4KB tile copy (256 threads x 16B). ASYNCcnt-tracked.
// ---------------------------------------------------------------------
#if __has_builtin(__builtin_amdgcn_global_load_async_to_lds_b128)
#define USE_ASYNC_LDS 1
#else
#define USE_ASYNC_LDS 0
#endif

__device__ __forceinline__ void stage_async_4k(const _Float16* __restrict__ gsrc,
                                               _Float16* __restrict__ ldst,
                                               int tid)
{
#if USE_ASYNC_LDS
    __builtin_amdgcn_global_load_async_to_lds_b128(
        (__attribute__((address_space(1))) v4i*)(gsrc + tid * 8),
        (__attribute__((address_space(3))) v4i*)(ldst + tid * 8),
        0, 0);
#else
    const uint4* s = (const uint4*)gsrc + tid;
    uint4*       d = (uint4*)ldst + tid;
    *d = *s;
#endif
}

__device__ __forceinline__ void wait_async_and_barrier()
{
#if USE_ASYNC_LDS
#if __has_builtin(__builtin_amdgcn_s_wait_asynccnt)
    __builtin_amdgcn_s_wait_asynccnt(0);
#else
    asm volatile("s_wait_asynccnt 0" ::: "memory");
#endif
#endif
    __syncthreads();
}

// ---------------------------------------------------------------------
// Fragment-ready packing.
// B tile (16x16 N per 32-K chunk): element (k, n) ->
//   lane = (n & 15) + 16*(k >> 4), slot = k & 15 ; tile = 512 halves.
// ---------------------------------------------------------------------
extern "C" __global__ void __launch_bounds__(256)
pack_w_leaf(const float* __restrict__ W_ioux, _Float16* __restrict__ Wl)
{
    int idx  = blockIdx.x * 256 + threadIdx.x;         // < 163840
    int slot = idx & 15;
    int lane = (idx >> 4) & 31;
    int j    = (idx >> 9) & 31;
    int kt   = idx >> 14;
    int nc   = j * 16 + (lane & 15);
    int k    = kt * 32 + (slot + ((lane >> 4) << 4));
    float v  = (nc < L_N && k < IN_DIM) ? W_ioux[nc * IN_DIM + k] : 0.f;
    Wl[idx]  = (_Float16)v;
}

extern "C" __global__ void __launch_bounds__(256)
pack_w_level(const float* __restrict__ W_iouh, const float* __restrict__ W_fh,
             _Float16* __restrict__ Wv)
{
    int idx   = blockIdx.x * 256 + threadIdx.x;        // < 102400
    int slot  = idx & 15;
    int lane  = (idx >> 4) & 31;
    int chunk = idx >> 9;                              // tile index 0..199
    int j     = chunk % V_TILES;
    int kt    = chunk / V_TILES;
    int nc    = j * 16 + (lane & 15);
    int k     = kt * 32 + (slot + ((lane >> 4) << 4));
    float v = 0.f;
    if (k < MEM) {
        if (nc < 450)      v = W_iouh[nc * MEM + k];
        else if (nc < V_N) v = W_fh[(nc - 450) * MEM + k];
    }
    Wv[idx] = (_Float16)v;
}

// A fragment-ready scatter: element (r, k) of a 16x32 tile ->
//   lane = r + 16*(k >> 4), slot = k & 15
#define A_LANE(r, k) ((r) + ((((k) >> 4) & 1) << 4))
#define A_SLOT(k)    ((k) & 15)

// Branch-free A stage: clamped index + select-zero (no exec divergence).
__device__ __forceinline__ void stage_a(const float* __restrict__ src_row_major,
                                        const long long* __restrict__ grow, // [2] rows for this thread's 2 elems
                                        int kdim, int k0, _Float16* __restrict__ lA,
                                        int tid)
{
#pragma unroll
    for (int it = 0; it < 2; ++it) {
        int e = tid + it * 256;
        int r = e >> 5, k = e & 31;
        int kg = k0 + k;
        int kc = (kg < kdim) ? kg : (kdim - 1);
        float v = src_row_major[grow[it] * kdim + kc];
        v = (kg < kdim) ? v : 0.f;
        lA[A_LANE(r, k) * 16 + A_SLOT(k)] = (_Float16)v;
    }
}

#define WMMA_F16(a, b, c) \
    __builtin_amdgcn_wmma_f32_16x16x32_f16(false, (a), false, (b), (short)0, (c), false, false)

// =====================================================================
// Leaf kernel: iou = x @ W_ioux^T + b_ioux + b_iouh ; gates ; c,h out.
// Block = 256 threads (8 waves), 16 leaf rows, 512 padded columns.
// Double-buffered LDS: async B copies + A stage overlap the WMMAs.
// =====================================================================
extern "C" __global__ void __launch_bounds__(256)
tree_lstm_leaf(const float* __restrict__ x,
               const _Float16* __restrict__ Wl,      // [10][32][512] frag-ready
               const float* __restrict__ b_ioux,
               const float* __restrict__ b_iouh,
               float* __restrict__ c_out,
               float* __restrict__ h_out)
{
    __shared__ __align__(32) unsigned char smem[2048 + 2 * L_TILES * 1024]; // 67584
    _Float16* lA0 = (_Float16*)smem;                    // [32][16]
    _Float16* lA1 = (_Float16*)(smem + 1024);
    _Float16* lB0 = (_Float16*)(smem + 2048);           // 32 KB
    _Float16* lB1 = (_Float16*)(smem + 2048 + L_TILES * 1024);
    float*    lO  = (float*)smem;                       // [16][512] overlay

    const int tid  = threadIdx.x;
    const int lane = tid & 31;
    const int wv   = __builtin_amdgcn_readfirstlane(tid >> 5);
    const int m    = lane & 15;
    const int hi   = lane >> 4;
    const long long row0 = (long long)blockIdx.x * 16;

    // the two A rows this thread stages (fixed across K chunks)
    long long grow[2];
#pragma unroll
    for (int it = 0; it < 2; ++it) grow[it] = row0 + ((tid + it * 256) >> 5);

    const v8f zero = {0.f, 0.f, 0.f, 0.f, 0.f, 0.f, 0.f, 0.f};
    v8f acc0 = zero, acc1 = zero, acc2 = zero, acc3 = zero;

    // prologue: chunk 0 in flight
#pragma unroll
    for (int c = 0; c < 8; ++c)
        stage_async_4k(Wl + c * 2048, lB0 + c * 2048, tid);
    stage_a(x, grow, IN_DIM, 0, lA0, tid);

    for (int kt = 0; kt < L_KT; ++kt) {
        wait_async_and_barrier();                  // chunk kt ready, kt-1 readers done
        const int cur = kt & 1;
        _Float16* bA = cur ? lA1 : lA0;
        _Float16* bB = cur ? lB1 : lB0;
        if (kt + 1 < L_KT) {                       // prefetch chunk kt+1 (other buffers)
            _Float16* nA = cur ? lA0 : lA1;
            _Float16* nB = cur ? lB0 : lB1;
            const _Float16* wsrc = Wl + (kt + 1) * (L_TILES * 512);
#pragma unroll
            for (int c = 0; c < 8; ++c)
                stage_async_4k(wsrc + c * 2048, nB + c * 2048, tid);
            stage_a(x, grow, IN_DIM, (kt + 1) * 32, nA, tid);
        }
        v16h af = *(const v16h*)(bA + lane * 16);
        const _Float16* bp = bB + (4 * wv) * 512 + lane * 16;
        v16h b0 = *(const v16h*)(bp);
        v16h b1 = *(const v16h*)(bp + 512);
        v16h b2 = *(const v16h*)(bp + 1024);
        v16h b3 = *(const v16h*)(bp + 1536);
        acc0 = WMMA_F16(af, b0, acc0);
        acc1 = WMMA_F16(af, b1, acc1);
        acc2 = WMMA_F16(af, b2, acc2);
        acc3 = WMMA_F16(af, b3, acc3);
    }
    __syncthreads();

    // dump accumulators: C layout -> row = v + 8*hi, col = 16*tile + (lane&15)
    {
        int nc0 = 16 * (4 * wv) + m;
#pragma unroll
        for (int v = 0; v < 8; ++v) {
            int r = (v + 8 * hi) * L_PITCH;
            lO[r + nc0]      = acc0[v];
            lO[r + nc0 + 16] = acc1[v];
            lO[r + nc0 + 32] = acc2[v];
            lO[r + nc0 + 48] = acc3[v];
        }
    }
    __syncthreads();

    // fused gates: c = sig(i)*tanh(u); h = sig(o)*tanh(c)
    for (int e = tid; e < 16 * MEM; e += 256) {
        int r = e / MEM, j = e - r * MEM;
        float iv = sigmoidf_(lO[r * L_PITCH + j]           + b_ioux[j]           + b_iouh[j]);
        float ov = sigmoidf_(lO[r * L_PITCH + MEM + j]     + b_ioux[MEM + j]     + b_iouh[MEM + j]);
        float uv = tanhf    (lO[r * L_PITCH + 2 * MEM + j] + b_ioux[2 * MEM + j] + b_iouh[2 * MEM + j]);
        float c = iv * uv;
        float h = ov * tanhf(c);
        long long g = (row0 + r) * MEM + j;
        c_out[g] = c;
        h_out[g] = h;
    }
}

// =====================================================================
// Level kernel: single GEMM over child rows with [W_iouh^T | W_fh^T],
// sibling-sum epilogue. Block = 8 parents (16 child rows), 640 cols.
// n is a power of two -> pass logN and use shifts (no 64-bit division).
// =====================================================================
extern "C" __global__ void __launch_bounds__(256)
tree_lstm_level(const float* __restrict__ c_in,
                const float* __restrict__ h_in,
                const _Float16* __restrict__ Wv,     // [5][40][512] frag-ready
                const float* __restrict__ b_iouh,
                const float* __restrict__ b_fh,
                float* __restrict__ c_out,
                float* __restrict__ h_out,
                int logN)                            // n = 1 << logN
{
    __shared__ __align__(32) unsigned char smem[2048 + 2 * V_TILES * 1024]; // 83968
    _Float16* lA0 = (_Float16*)smem;
    _Float16* lA1 = (_Float16*)(smem + 1024);
    _Float16* lB0 = (_Float16*)(smem + 2048);           // 40 KB
    _Float16* lB1 = (_Float16*)(smem + 2048 + V_TILES * 1024);
    float*    lO  = (float*)smem;                       // [16][640] overlay

    const int tid  = threadIdx.x;
    const int lane = tid & 31;
    const int wv   = __builtin_amdgcn_readfirstlane(tid >> 5);
    const int m    = lane & 15;
    const int hi   = lane >> 4;
    const int logHalf = logN - 1;
    const long long halfMask = (1LL << logHalf) - 1;
    const long long P0 = (long long)blockIdx.x * 8;

    auto childRow = [&](int r) -> long long {
        long long P    = P0 + (r >> 1);
        long long tree = P >> logHalf;
        long long idx  = P & halfMask;
        return (tree << logN) + 2 * idx + (r & 1);
    };

    long long grow[2];
#pragma unroll
    for (int it = 0; it < 2; ++it) grow[it] = childRow((tid + it * 256) >> 5);

    const v8f zero = {0.f, 0.f, 0.f, 0.f, 0.f, 0.f, 0.f, 0.f};
    v8f acc0 = zero, acc1 = zero, acc2 = zero, acc3 = zero, acc4 = zero;

#pragma unroll
    for (int c = 0; c < 10; ++c)
        stage_async_4k(Wv + c * 2048, lB0 + c * 2048, tid);
    stage_a(h_in, grow, MEM, 0, lA0, tid);

    for (int kt = 0; kt < V_KT; ++kt) {
        wait_async_and_barrier();
        const int cur = kt & 1;
        _Float16* bA = cur ? lA1 : lA0;
        _Float16* bB = cur ? lB1 : lB0;
        if (kt + 1 < V_KT) {
            _Float16* nA = cur ? lA0 : lA1;
            _Float16* nB = cur ? lB0 : lB1;
            const _Float16* wsrc = Wv + (kt + 1) * (V_TILES * 512);
#pragma unroll
            for (int c = 0; c < 10; ++c)
                stage_async_4k(wsrc + c * 2048, nB + c * 2048, tid);
            stage_a(h_in, grow, MEM, (kt + 1) * 32, nA, tid);
        }
        v16h af = *(const v16h*)(bA + lane * 16);
        const _Float16* bp = bB + (5 * wv) * 512 + lane * 16;
        v16h b0 = *(const v16h*)(bp);
        v16h b1 = *(const v16h*)(bp + 512);
        v16h b2 = *(const v16h*)(bp + 1024);
        v16h b3 = *(const v16h*)(bp + 1536);
        v16h b4 = *(const v16h*)(bp + 2048);
        acc0 = WMMA_F16(af, b0, acc0);
        acc1 = WMMA_F16(af, b1, acc1);
        acc2 = WMMA_F16(af, b2, acc2);
        acc3 = WMMA_F16(af, b3, acc3);
        acc4 = WMMA_F16(af, b4, acc4);
    }
    __syncthreads();

    {
        int nc0 = 16 * (5 * wv) + m;
#pragma unroll
        for (int v = 0; v < 8; ++v) {
            int r = (v + 8 * hi) * V_PITCH;
            lO[r + nc0]      = acc0[v];
            lO[r + nc0 + 16] = acc1[v];
            lO[r + nc0 + 32] = acc2[v];
            lO[r + nc0 + 48] = acc3[v];
            lO[r + nc0 + 64] = acc4[v];
        }
    }
    __syncthreads();

    // epilogue: sibling-sum -> i,o,u ; per-child forget gates ; fold cells
    for (int e = tid; e < 8 * MEM; e += 256) {
        int p = e / MEM, j = e - p * MEM;
        int r0 = 2 * p, r1 = r0 + 1;
        float iv = sigmoidf_(lO[r0 * V_PITCH + j]           + lO[r1 * V_PITCH + j]           + b_iouh[j]);
        float ov = sigmoidf_(lO[r0 * V_PITCH + MEM + j]     + lO[r1 * V_PITCH + MEM + j]     + b_iouh[MEM + j]);
        float uv = tanhf    (lO[r0 * V_PITCH + 2 * MEM + j] + lO[r1 * V_PITCH + 2 * MEM + j] + b_iouh[2 * MEM + j]);
        float f0 = sigmoidf_(lO[r0 * V_PITCH + 450 + j] + b_fh[j]);
        float f1 = sigmoidf_(lO[r1 * V_PITCH + 450 + j] + b_fh[j]);
        long long g0 = childRow(r0) * MEM + j;
        long long g1 = childRow(r1) * MEM + j;
        float c = iv * uv + f0 * c_in[g0] + f1 * c_in[g1];
        float h = ov * tanhf(c);
        long long og = (P0 + p) * MEM + j;   // [T][n/2][MEM] flat == parent idx * MEM
        c_out[og] = c;
        h_out[og] = h;
    }
}

// =====================================================================
// Host launcher
// =====================================================================
extern "C" void kernel_launch(void* const* d_in, const int* in_sizes, int n_in,
                              void* d_out, int out_size, void* d_ws, size_t ws_size,
                              hipStream_t stream)
{
    const float* x      = (const float*)d_in[0];
    const float* W_ioux = (const float*)d_in[1];
    const float* b_ioux = (const float*)d_in[2];
    const float* W_iouh = (const float*)d_in[3];
    const float* b_iouh = (const float*)d_in[4];
    // d_in[5] = W_fx, d_in[6] = b_fx: unused (leaf forget gate multiplies zero
    // child state, so it never reaches the output).
    const float* W_fh   = (const float*)d_in[7];
    const float* b_fh   = (const float*)d_in[8];
    float* out = (float*)d_out;
    float* ws  = (float*)d_ws;

    const long long SZ1 = (long long)T_DIM * LEAVES * MEM;  // full level (c or h)
    const long long SZ2 = SZ1 / 2;
    float* Ac = ws;
    float* Ah = Ac + SZ1;
    float* Bc = Ah + SZ1;
    float* Bh = Bc + SZ2;
    _Float16* Wl = (_Float16*)(Bh + SZ2);
    _Float16* Wv = Wl + WL_ELEMS;

    // pack weights into fragment-ready f16 (L2-resident, reused by all blocks)
    pack_w_leaf <<<WL_ELEMS / 256, 256, 0, stream>>>(W_ioux, Wl);
    pack_w_level<<<WV_ELEMS / 256, 256, 0, stream>>>(W_iouh, W_fh, Wv);

    // leaves: 262144 rows / 16 per block
    tree_lstm_leaf<<<(T_DIM * LEAVES) / 16, 256, 0, stream>>>(
        x, Wl, b_ioux, b_iouh, Ac, Ah);

    // reduction levels n = 256 .. 4 (ping-pong), final n = 2 writes d_out
    const float* cin = Ac;
    const float* hin = Ah;
    float* co = Bc;
    float* ho = Bh;
    for (int logN = 8; logN >= 2; --logN) {
        int blocks = (T_DIM * (1 << (logN - 1))) / 8;
        tree_lstm_level<<<blocks, 256, 0, stream>>>(
            cin, hin, Wv, b_iouh, b_fh, co, ho, logN);
        const float* tc = cin; const float* th = hin;
        cin = co; hin = ho;
        co = (float*)tc; ho = (float*)th;
    }
    // n = 2: 1024 parents -> root child's (c, h) == reference output [2, T, MEM]
    tree_lstm_level<<<T_DIM / 8, 256, 0, stream>>>(
        cin, hin, Wv, b_iouh, b_fh,
        out, out + (long long)T_DIM * MEM, 1);
}
